// RoadGM_48284022341689
// MI455X (gfx1250) — matrix operand order repaired and checked by reading, not compiled, not run on hardware
//
#include <hip/hip_runtime.h>
#include <hip/hip_bf16.h>

#define NNODES 50000
#define NEDGES 800000
#define DM 256

typedef __attribute__((ext_vector_type(16))) __bf16 v16bf;
typedef __attribute__((ext_vector_type(8)))  float  v8f;
typedef int v4i_vs __attribute__((vector_size(16)));           // matches builtin param type
typedef __attribute__((address_space(1))) v4i_vs* g_v4i_ptr;   // global int4*
typedef __attribute__((address_space(3))) v4i_vs* l_v4i_ptr;   // LDS int4*

// ---------- helpers ----------
__device__ __forceinline__ unsigned short f32_to_bf16_bits(float f) {
    unsigned u = __float_as_uint(f);
    unsigned r = 0x7FFFu + ((u >> 16) & 1u);   // round-to-nearest-even
    return (unsigned short)((u + r) >> 16);
}
// monotone float<->uint map for atomicMax on floats
__device__ __forceinline__ unsigned fenc(float f) {
    unsigned u = __float_as_uint(f);
    return (u & 0x80000000u) ? ~u : (u | 0x80000000u);
}
__device__ __forceinline__ float fdec(unsigned e) {
    return __uint_as_float((e & 0x80000000u) ? (e & 0x7FFFFFFFu) : ~e);
}

// =====================================================================
// Fragment-major bf16 conversion.
// A-fragment layout (v_wmma_* 16x16x32, ISA 7.12.2): element j of lane L in
// tile (mt,kb):  m = mt*16 + (L&15),
//                k = kb*32 + (L>>4)*8 + ((j>>1)>>2)*16 + (((j>>1)&3)*2) + (j&1)
// B-fragment layout: element j of lane L in tile (kb,nt):
//                col = nt*16 + (L&15),  k = kb*32 + (L>>4)*8 + (j>>3)*16 + (j&7)
// Each lane's 16 elements are stored contiguously (32 bytes) so the GEMM
// loads one v16bf per fragment with b128 loads.
// =====================================================================
__global__ __launch_bounds__(256) void perm_cvt_A_kernel(const float* __restrict__ in,
                                                         unsigned short* __restrict__ out,
                                                         int K, size_t total) {
    size_t i = (size_t)blockIdx.x * blockDim.x + threadIdx.x;
    if (i >= total) return;
    int tile = (int)(i >> 9);
    int w    = (int)(i & 511);
    int lane = w >> 4, j = w & 15;
    int kblocks = K >> 5;
    int mt = tile / kblocks, kb = tile % kblocks;
    int m = (mt << 4) + (lane & 15);
    int v = j >> 1;
    int k = (kb << 5) + ((lane >> 4) << 3) + ((v >> 2) << 4) + ((v & 3) << 1) + (j & 1);
    out[i] = f32_to_bf16_bits(in[(size_t)m * K + k]);
}

__global__ __launch_bounds__(256) void perm_cvt_B_kernel(const float* __restrict__ in,
                                                         unsigned short* __restrict__ out,
                                                         size_t total) {
    size_t i = (size_t)blockIdx.x * blockDim.x + threadIdx.x;
    if (i >= total) return;
    int tile = (int)(i >> 9);
    int w    = (int)(i & 511);
    int lane = w >> 4, j = w & 15;
    int kb = tile >> 4, nt = tile & 15;
    int col = (nt << 4) + (lane & 15);
    int k = (kb << 5) + ((lane >> 4) << 3) + ((j >> 3) << 4) + (j & 7);
    out[i] = f32_to_bf16_bits(in[(size_t)k * DM + col]);
}

// ---------- WMMA GEMM:  C[M x 256] = A[M x K] * B[K x 256]  (bf16 in, f32 out) ----
// block = 256 threads = 8 waves; block covers 16 rows x 256 cols; wave w does
// n-tiles 2w, 2w+1.  A tile (1 KB) staged in LDS once per k-step via the CDNA5
// async global->LDS path, B fragments loaded contiguously from global.
__global__ __launch_bounds__(256) void gemm_bf16_wmma(const unsigned short* __restrict__ A,
                                                      const unsigned short* __restrict__ B,
                                                      float* __restrict__ C, int K) {
    __shared__ __align__(32) unsigned short sA[512];   // one 16x32 bf16 tile
    const int tid  = threadIdx.x;
    const int wave = tid >> 5;
    const int lane = tid & 31;
    const int half = lane >> 4;
    const int l16  = lane & 15;
    const int m0   = blockIdx.x << 4;
    const int kblocks = K >> 5;

    const unsigned short* Atile = A + (size_t)blockIdx.x * kblocks * 512;
    v8f acc0 = {}; v8f acc1 = {};

    for (int kb = 0; kb < kblocks; ++kb) {
#if __has_builtin(__builtin_amdgcn_global_load_async_to_lds_b128)
        if (tid < 64) {
            __builtin_amdgcn_global_load_async_to_lds_b128(
                (g_v4i_ptr)(Atile + (size_t)kb * 512 + tid * 8),
                (l_v4i_ptr)(sA + tid * 8),
                0, 0);
        }
#if __has_builtin(__builtin_amdgcn_s_wait_asynccnt)
        __builtin_amdgcn_s_wait_asynccnt(0);
#else
        asm volatile("s_wait_asynccnt 0" ::: "memory");
#endif
#else
        // fallback: cooperative dword staging
        ((unsigned*)sA)[tid] = ((const unsigned*)(Atile + (size_t)kb * 512))[tid];
#endif
        __syncthreads();

        v16bf a  = *(const v16bf*)(sA + lane * 16);                 // 2x ds_load_b128
        const unsigned short* bp = B + ((size_t)kb * 16 + (wave << 1)) * 512 + lane * 16;
        v16bf b0 = *(const v16bf*)(bp);                              // 2x global_load_b128
        v16bf b1 = *(const v16bf*)(bp + 512);

        acc0 = __builtin_amdgcn_wmma_f32_16x16x32_bf16(false, a, false, b0, (short)0, acc0, false, false);
        acc1 = __builtin_amdgcn_wmma_f32_16x16x32_bf16(false, a, false, b1, (short)0, acc1, false, false);
        __syncthreads();
    }

    // C layout: VGPR r -> M = m0 + r + 8*half, N = lane%16.  M % 16 == 0: no guards.
    const int n0 = wave << 5;
#pragma unroll
    for (int r = 0; r < 8; ++r) {
        size_t row = (size_t)(m0 + r + (half << 3)) * DM;
        C[row + n0 + l16]      = acc0[r];
        C[row + n0 + 16 + l16] = acc1[r];
    }
}

// ---------- per-node attention scores: s[n,h] = sum_f proj[n,h*F+f] * a[h*F+f] ----------
__global__ __launch_bounds__(256) void attn_scores_kernel(const float* __restrict__ proj,
                                                          const float* __restrict__ a_src,
                                                          const float* __restrict__ a_tgt,
                                                          float* __restrict__ ssrc,
                                                          float* __restrict__ stgt,
                                                          int NH, int F) {
    int n = blockIdx.x, t = threadIdx.x;
    __shared__ float sh1[256], sh2[256];
    float p = proj[(size_t)n * DM + t];
    sh1[t] = p * a_src[t];
    sh2[t] = p * a_tgt[t];
    __syncthreads();
    for (int s = F >> 1; s > 0; s >>= 1) {
        if ((t & (F - 1)) < s) { sh1[t] += sh1[t + s]; sh2[t] += sh2[t + s]; }
        __syncthreads();
    }
    if ((t & (F - 1)) == 0) {
        ssrc[(size_t)n * NH + t / F] = sh1[t];
        stgt[(size_t)n * NH + t / F] = sh2[t];
    }
}

// ---------- c_tp[h] = sum_f Wtp[h*F+f] * a_tp[h*F+f] ----------
__global__ __launch_bounds__(256) void ctp_kernel(const float* __restrict__ Wtp,
                                                  const float* __restrict__ a_tp,
                                                  float* __restrict__ ctp, int NH, int F) {
    int t = threadIdx.x;
    __shared__ float sh[256];
    sh[t] = Wtp[t] * a_tp[t];
    __syncthreads();
    for (int s = F >> 1; s > 0; s >>= 1) {
        if ((t & (F - 1)) < s) sh[t] += sh[t + s];
        __syncthreads();
    }
    if ((t & (F - 1)) == 0) ctp[t / F] = sh[t];
}

// ---------- edge pass 1: leaky-relu score + segment max (over tgt) ----------
__global__ __launch_bounds__(256) void edge_scores_kernel(const int* __restrict__ src,
                                                          const int* __restrict__ tgt,
                                                          const float* __restrict__ eprob,
                                                          const float* __restrict__ ssrc,
                                                          const float* __restrict__ stgt,
                                                          const float* __restrict__ ctp,
                                                          float* __restrict__ ew,
                                                          unsigned* __restrict__ nmax,
                                                          int E, int NH) {
    size_t i = (size_t)blockIdx.x * blockDim.x + threadIdx.x;
    if (i >= (size_t)E * NH) return;
    int e = (int)(i / NH), h = (int)(i % NH);
    int s = src[e], t = tgt[e];
    float sc = ssrc[(size_t)s * NH + h] + stgt[(size_t)t * NH + h] + eprob[e] * ctp[h];
    sc = sc > 0.f ? sc : 0.2f * sc;          // leaky_relu(0.2)
    ew[i] = sc;
    atomicMax(&nmax[(size_t)t * NH + h], fenc(sc));
}

// ---------- edge pass 2: w = exp(sc - max), segment sum ----------
__global__ __launch_bounds__(256) void edge_softmax_kernel(const int* __restrict__ tgt,
                                                           float* __restrict__ ew,
                                                           const unsigned* __restrict__ nmax,
                                                           float* __restrict__ denom,
                                                           int E, int NH) {
    size_t i = (size_t)blockIdx.x * blockDim.x + threadIdx.x;
    if (i >= (size_t)E * NH) return;
    int e = (int)(i / NH), h = (int)(i % NH);
    int t = tgt[e];
    float w = __expf(ew[i] - fdec(nmax[(size_t)t * NH + h]));
    ew[i] = w;
    atomicAdd(&denom[(size_t)t * NH + h], w);
}

// ---------- edge pass 3: agg[tgt, c] += proj[src, c] * alpha ----------
__global__ __launch_bounds__(256) void edge_scatter_kernel(const int* __restrict__ src,
                                                           const int* __restrict__ tgt,
                                                           const float* __restrict__ ew,
                                                           const float* __restrict__ denom,
                                                           const float* __restrict__ proj,
                                                           float* __restrict__ agg,
                                                           int E, int NH, int F) {
    size_t i = (size_t)blockIdx.x * blockDim.x + threadIdx.x;
    if (i >= (size_t)E * DM) return;
    int e = (int)(i >> 8), c = (int)(i & 255);
    int h = c / F;
    int s = src[e], t = tgt[e];
    float alpha = ew[(size_t)e * NH + h] / (denom[(size_t)t * NH + h] + 1e-16f);
    atomicAdd(&agg[(size_t)t * DM + c], proj[(size_t)s * DM + c] * alpha);
}

// ---------- finalize: skip-add + bias + ELU (+ LayerNorm), in place over agg ----------
__global__ __launch_bounds__(256) void finalize_kernel(float* __restrict__ agg,
                                                       const float* __restrict__ skipb,   // nullable
                                                       const float* __restrict__ h_ident, // identity skip
                                                       const float* __restrict__ bias,
                                                       const float* __restrict__ ln_g,
                                                       const float* __restrict__ ln_b,
                                                       int do_ln) {
    int n = blockIdx.x, t = threadIdx.x;
    size_t idx = (size_t)n * DM + t;
    float v = agg[idx];
    v += skipb ? skipb[idx] : h_ident[idx];
    v += bias[t];
    v = v > 0.f ? v : (__expf(v) - 1.f);     // elu
    if (do_ln) {
        __shared__ float sh[256];
        sh[t] = v; __syncthreads();
        for (int s = 128; s > 0; s >>= 1) { if (t < s) sh[t] += sh[t + s]; __syncthreads(); }
        float mu = sh[0] * (1.f / 256.f); __syncthreads();
        float d = v - mu;
        sh[t] = d * d; __syncthreads();
        for (int s = 128; s > 0; s >>= 1) { if (t < s) sh[t] += sh[t + s]; __syncthreads(); }
        float var = sh[0] * (1.f / 256.f);
        v = d * rsqrtf(var + 1e-5f) * ln_g[t] + ln_b[t];
    }
    agg[idx] = v;
}

// ---------- final gather out[b,s,:] = h[x[b,s],:] ----------
__global__ __launch_bounds__(256) void gather_kernel(const float* __restrict__ h,
                                                     const int* __restrict__ x,
                                                     float* __restrict__ out, size_t total) {
    size_t i = (size_t)blockIdx.x * blockDim.x + threadIdx.x;
    if (i >= total) return;
    int row = x[i >> 8];
    out[i] = h[(size_t)row * DM + (i & 255)];
}

// ---------- host-side layer driver ----------
struct Scratch {
    float *proj, *skipb, *ssrc, *stgt, *denom, *ew, *ctp;
    unsigned *nmax;
    unsigned short *abf, *wbf;
};

static void run_layer(const float* h_in, int K,
                      const float* W, const float* Wskip,   // Wskip==nullptr -> identity skip
                      const float* a_src, const float* a_tgt,
                      const float* Wtp, const float* a_tp,
                      const float* bias, const float* ln_g, const float* ln_b, int do_ln,
                      int NH, int F,
                      const int* src, const int* tgt, const float* eprob,
                      float* agg, const Scratch& S, hipStream_t stream) {
    const size_t N = NNODES, E = NEDGES;

    size_t na = N * (size_t)K;
    perm_cvt_A_kernel<<<(unsigned)((na + 255) / 256), 256, 0, stream>>>(h_in, S.abf, K, na);
    size_t nw = (size_t)K * DM;
    perm_cvt_B_kernel<<<(unsigned)((nw + 255) / 256), 256, 0, stream>>>(W, S.wbf, nw);
    gemm_bf16_wmma<<<(unsigned)(N / 16), 256, 0, stream>>>(S.abf, S.wbf, S.proj, K);
    if (Wskip) {
        perm_cvt_B_kernel<<<(unsigned)((nw + 255) / 256), 256, 0, stream>>>(Wskip, S.wbf, nw);
        gemm_bf16_wmma<<<(unsigned)(N / 16), 256, 0, stream>>>(S.abf, S.wbf, S.skipb, K);
    }

    attn_scores_kernel<<<(unsigned)N, 256, 0, stream>>>(S.proj, a_src, a_tgt, S.ssrc, S.stgt, NH, F);
    ctp_kernel<<<1, 256, 0, stream>>>(Wtp, a_tp, S.ctp, NH, F);

    (void)hipMemsetAsync(agg,     0, N * DM * sizeof(float), stream);
    (void)hipMemsetAsync(S.nmax,  0, N * (size_t)NH * sizeof(unsigned), stream);
    (void)hipMemsetAsync(S.denom, 0, N * (size_t)NH * sizeof(float), stream);

    size_t w1 = E * (size_t)NH;
    edge_scores_kernel<<<(unsigned)((w1 + 255) / 256), 256, 0, stream>>>(
        src, tgt, eprob, S.ssrc, S.stgt, S.ctp, S.ew, S.nmax, (int)E, NH);
    edge_softmax_kernel<<<(unsigned)((w1 + 255) / 256), 256, 0, stream>>>(
        tgt, S.ew, S.nmax, S.denom, (int)E, NH);
    size_t w3 = E * (size_t)DM;
    edge_scatter_kernel<<<(unsigned)((w3 + 255) / 256), 256, 0, stream>>>(
        src, tgt, S.ew, S.denom, S.proj, agg, (int)E, NH, F);

    finalize_kernel<<<(unsigned)N, 256, 0, stream>>>(
        agg, Wskip ? S.skipb : (const float*)nullptr, h_in, bias, ln_g, ln_b, do_ln);
}

extern "C" void kernel_launch(void* const* d_in, const int* in_sizes, int n_in,
                              void* d_out, int out_size, void* d_ws, size_t ws_size,
                              hipStream_t stream) {
    const float* node_features = (const float*)d_in[0];
    const int*   edge_index    = (const int*)d_in[1];
    const float* edge_prob     = (const float*)d_in[2];
    const int*   x             = (const int*)d_in[3];
    const float* W0     = (const float*)d_in[4];
    const float* a_src0 = (const float*)d_in[5];
    const float* a_tgt0 = (const float*)d_in[6];
    const float* Wtp0   = (const float*)d_in[7];
    const float* a_tp0  = (const float*)d_in[8];
    const float* Wskip0 = (const float*)d_in[9];
    const float* b0     = (const float*)d_in[10];
    const float* W1     = (const float*)d_in[11];
    const float* a_src1 = (const float*)d_in[12];
    const float* a_tgt1 = (const float*)d_in[13];
    const float* Wtp1   = (const float*)d_in[14];
    const float* a_tp1  = (const float*)d_in[15];
    const float* Wskip1 = (const float*)d_in[16];
    const float* b1     = (const float*)d_in[17];
    const float* ln1_g  = (const float*)d_in[18];
    const float* ln1_b  = (const float*)d_in[19];
    const float* W2     = (const float*)d_in[20];
    const float* a_src2 = (const float*)d_in[21];
    const float* a_tgt2 = (const float*)d_in[22];
    const float* Wtp2   = (const float*)d_in[23];
    const float* a_tp2  = (const float*)d_in[24];
    const float* b2     = (const float*)d_in[25];
    const float* ln2_g  = (const float*)d_in[26];
    const float* ln2_b  = (const float*)d_in[27];

    const int* src = edge_index;
    const int* tgt = edge_index + NEDGES;

    // carve workspace
    char* p = (char*)d_ws;
    auto carve = [&](size_t bytes) -> void* {
        void* r = (void*)p;
        p += (bytes + 255) & ~(size_t)255;
        return r;
    };
    const size_t N = NNODES, E = NEDGES;
    float* bufA  = (float*)carve(N * DM * 4);
    float* bufB  = (float*)carve(N * DM * 4);
    Scratch S;
    S.proj  = (float*)carve(N * DM * 4);
    S.skipb = (float*)carve(N * DM * 4);
    S.ssrc  = (float*)carve(N * 16 * 4);
    S.stgt  = (float*)carve(N * 16 * 4);
    S.nmax  = (unsigned*)carve(N * 16 * 4);
    S.denom = (float*)carve(N * 16 * 4);
    S.ew    = (float*)carve(E * 16 * 4);
    S.abf   = (unsigned short*)carve(N * DM * 2);
    S.wbf   = (unsigned short*)carve((size_t)DM * DM * 2);
    S.ctp   = (float*)carve(64);

    // layer 0: 128 -> 16x16 concat, skip GEMM, no LN
    run_layer(node_features, 128, W0, Wskip0, a_src0, a_tgt0, Wtp0, a_tp0,
              b0, nullptr, nullptr, /*do_ln=*/0, /*NH=*/16, /*F=*/16,
              src, tgt, edge_prob, bufA, S, stream);
    // layer 1: 256 -> 16x16 concat, skip GEMM, LN
    run_layer(bufA, 256, W1, Wskip1, a_src1, a_tgt1, Wtp1, a_tp1,
              b1, ln1_g, ln1_b, /*do_ln=*/1, 16, 16,
              src, tgt, edge_prob, bufB, S, stream);
    // layer 2: 256 -> 1x256 avg (NH=1 -> identity mean), identity skip, LN
    run_layer(bufB, 256, W2, nullptr, a_src2, a_tgt2, Wtp2, a_tp2,
              b2, ln2_g, ln2_b, /*do_ln=*/1, 1, 256,
              src, tgt, edge_prob, bufA, S, stream);

    size_t total = (size_t)out_size;   // 64*128*256
    gather_kernel<<<(unsigned)((total + 255) / 256), 256, 0, stream>>>(bufA, x, (float*)d_out, total);
}
